// BigramLanguageModel_56951266345194
// MI455X (gfx1250) — matrix operand order repaired and checked by reading, not compiled, not run on hardware
//
#include <hip/hip_runtime.h>
#include <stdint.h>

// ---------------------------------------------------------------------------
// CDNA5 (gfx1250) GPT forward pass.
// - All GEMMs on v_wmma_f32_16x16x32_bf16 (wave32, fp32 accum).
// - GEMM tiles staged with global_load_async_to_lds_b128 (ASYNCcnt path),
//   DOUBLE-BUFFERED: tile k+1 streams in while tile k is computed; the
//   in-order ASYNCcnt lets us wait "<=4" (the 4 just-issued copies) so the
//   previous tile is guaranteed resident with zero idle wait.
// - Weights packed transposed [N][K] so both tiles are contiguous slabs.
// ---------------------------------------------------------------------------

typedef __attribute__((ext_vector_type(16))) __bf16 v16bf;
typedef __attribute__((ext_vector_type(8)))  float  v8f;

union frag_ab { v16bf v; uint32_t u[8]; };

__device__ __forceinline__ v8f wmma_bf16(const frag_ab& a, const frag_ab& b, v8f c) {
  return __builtin_amdgcn_wmma_f32_16x16x32_bf16(
      false, a.v, false, b.v, (short)0, c, false, false);
}

__device__ __forceinline__ uint16_t f2bf(float f) {
  union { float f; uint32_t u; } cv; cv.f = f;
  uint32_t u = cv.u;
  return (uint16_t)((u + 0x7FFFu + ((u >> 16) & 1u)) >> 16);
}

__device__ __forceinline__ v8f v8f_zero() {
  v8f z; for (int i = 0; i < 8; ++i) z[i] = 0.f; return z;
}

// Per-lane async copy of 16 bytes global -> LDS (ASYNCcnt-tracked).
__device__ __forceinline__ void async_copy_b128(uint32_t lds_addr, uint64_t gaddr) {
  asm volatile("global_load_async_to_lds_b128 %0, %1, off"
               :: "v"(lds_addr), "v"(gaddr) : "memory");
}
template <int N>
__device__ __forceinline__ void async_wait_le() {
  asm volatile("s_wait_asynccnt %0" :: "i"(N) : "memory");
}

// ---------------------------------------------------------------------------
__global__ __launch_bounds__(256)
void embed_kernel(const float* __restrict__ te, const float* __restrict__ pe,
                  const int* __restrict__ ctx, float* __restrict__ x) {
  const int tok = blockIdx.x, t = threadIdx.x;
  const int c = ctx[tok], pos = tok & 1023;
  const float* tr = te + (size_t)c * 1024;
  const float* pr = pe + (size_t)pos * 1024;
  float* xr = x + (size_t)tok * 1024;
  for (int j = 0; j < 4; ++j) { int i = t + j * 256; xr[i] = tr[i] + pr[i]; }
}

// ---------------------------------------------------------------------------
__global__ __launch_bounds__(256)
void ln_kernel(const float* __restrict__ x, const float* __restrict__ g,
               const float* __restrict__ bb, uint16_t* __restrict__ out) {
  __shared__ float red[256];
  const int row = blockIdx.x, t = threadIdx.x;
  const float* xr = x + (size_t)row * 1024;
  float v[4], s = 0.f;
  for (int j = 0; j < 4; ++j) { v[j] = xr[t + j * 256]; s += v[j]; }
  red[t] = s; __syncthreads();
  for (int o = 128; o > 0; o >>= 1) { if (t < o) red[t] += red[t + o]; __syncthreads(); }
  const float mean = red[0] * (1.f / 1024.f);
  __syncthreads();
  float q = 0.f;
  for (int j = 0; j < 4; ++j) { float d = v[j] - mean; q += d * d; }
  red[t] = q; __syncthreads();
  for (int o = 128; o > 0; o >>= 1) { if (t < o) red[t] += red[t + o]; __syncthreads(); }
  const float inv = rsqrtf(red[0] * (1.f / 1024.f) + 1e-5f);
  uint16_t* orow = out + (size_t)row * 1024;
  for (int j = 0; j < 4; ++j) {
    int i = t + j * 256;
    orow[i] = f2bf((v[j] - mean) * inv * g[i] + bb[i]);
  }
}

// ---------------------------------------------------------------------------
// Weight converters fp32 -> bf16, packed TRANSPOSED: dst[n][k] = src[k][n].
// ---------------------------------------------------------------------------
__global__ __launch_bounds__(256)
void cvt_t_kernel(const float* __restrict__ src, uint16_t* __restrict__ dst,
                  int kshift, int N) {
  const int i = blockIdx.x * 256 + threadIdx.x;   // output index over N*K
  const int K = 1 << kshift;
  const int k = i & (K - 1), n = i >> kshift;
  dst[i] = f2bf(src[(size_t)k * N + n]);
}

// Wq[h,d,e] (H=16,D=1024,HS=64) -> transposed packed [n=h*64+e][d]
__global__ __launch_bounds__(256)
void cvt_qkv_t_kernel(const float* __restrict__ src, uint16_t* __restrict__ dst) {
  const int i = blockIdx.x * 256 + threadIdx.x;   // over 1<<20, output index
  const int d = i & 1023, n = i >> 10, h = n >> 6, e = n & 63;
  dst[i] = f2bf(src[(size_t)h * 65536 + (size_t)d * 64 + e]);
}

// v[token, h*64+e] -> vT[(b*16+h)*64+e, t]
__global__ __launch_bounds__(256)
void transpose_v_kernel(const uint16_t* __restrict__ v, uint16_t* __restrict__ vT) {
  const int n = blockIdx.x * 256 + threadIdx.x;   // over 4*16*64*1024
  const int t = n & 1023, rest = n >> 10, e = rest & 63, bh = rest >> 6;
  const int b = bh >> 4, h = bh & 15;
  vT[n] = v[((size_t)(b * 1024 + t)) * 1024 + h * 64 + e];
}

// ---------------------------------------------------------------------------
// Tiled bf16 WMMA GEMM: C[M,N] = A[M,K] * Bt[N,K]^T (+bias)(+relu)(+resid)
// Block = 256 thr = 8 waves; tile 128x128x32; each wave owns a 16x128 strip.
// Double-buffered async-to-LDS pipeline.
// LDS row stride 40 halves = 80 B: 16B aligned for b128, 20-bank skew.
// ---------------------------------------------------------------------------
__global__ __launch_bounds__(256)
void gemm_bf16_kernel(const uint16_t* __restrict__ A, const uint16_t* __restrict__ Bt,
                      const float* __restrict__ bias, const float* __restrict__ resid,
                      float* __restrict__ outF, uint16_t* __restrict__ outB,
                      int M, int N, int K, int relu) {
  __shared__ uint16_t As[2][128][40];   // [buf][m][k]
  __shared__ uint16_t Bs[2][128][40];   // [buf][n][k]
  const int t = threadIdx.x;
  const int lane = t & 31, wave = t >> 5;
  const int half = lane >> 4, l16 = lane & 15;
  const int bm = blockIdx.y * 128, bn = blockIdx.x * 128;

  const uint32_t lds_a0 = (uint32_t)(uintptr_t)&As[0][0][0];
  const uint32_t lds_b0 = (uint32_t)(uintptr_t)&Bs[0][0][0];
  const uint64_t ga_base = (uint64_t)(uintptr_t)A + 2ull * (size_t)bm * K;
  const uint64_t gb_base = (uint64_t)(uintptr_t)Bt + 2ull * (size_t)bn * K;

  // Issue the 4 async copies (2 chunks x {A,B}) for one 128x32 tile pair.
  const int c_row = t >> 2, c_ch = t & 3;            // 4 x 16B chunks per row
  auto issue_tiles = [&](int k0, int buf) {
    const uint32_t lbase = (uint32_t)(buf * 10240);
    for (int j = 0; j < 2; ++j) {
      const int row = c_row + j * 64;
      const uint32_t loff = lbase + (uint32_t)(row * 80 + c_ch * 16);
      const uint64_t goff = 2ull * ((size_t)row * K + k0) + (uint64_t)(c_ch * 16);
      async_copy_b128(lds_a0 + loff, ga_base + goff);
      async_copy_b128(lds_b0 + loff, gb_base + goff);
    }
  };

  v8f c[8];
  for (int nt = 0; nt < 8; ++nt) c[nt] = v8f_zero();

  issue_tiles(0, 0);
  int cur = 0;
  for (int k0 = 0; k0 < K; k0 += 32) {
    const bool has_next = (k0 + 32) < K;
    if (has_next) {
      issue_tiles(k0 + 32, cur ^ 1);
      async_wait_le<4>();      // the 4 just-issued stay outstanding;
    } else {                   // everything older (current tile) has landed
      async_wait_le<0>();
    }
    __syncthreads();

    frag_ab a; // A 16x32: lane = M row, K pairs split by lane-half
    {
      const uint16_t* arow = &As[cur][wave * 16 + l16][0];
      const int kb8 = half * 8;
      for (int i = 0; i < 4; ++i) {
        a.u[i]     = *(const uint32_t*)&arow[kb8 + 2 * i];
        a.u[4 + i] = *(const uint32_t*)&arow[16 + kb8 + 2 * i];
      }
    }
    for (int nt = 0; nt < 8; ++nt) {
      frag_ab bf; // B 32x16: lane = N col, K pairs split by lane-half
      const uint16_t* brow = &Bs[cur][nt * 16 + l16][0];
      const int kb8b = half * 8;
      for (int i = 0; i < 8; ++i)
        bf.u[i] = *(const uint32_t*)&brow[(i >> 2) * 16 + kb8b + 2 * (i & 3)];
      c[nt] = wmma_bf16(a, bf, c[nt]);
    }
    __syncthreads();           // next iter overwrites buf (cur^1)^1 = cur
    cur ^= 1;
  }

  for (int nt = 0; nt < 8; ++nt) {
    for (int r = 0; r < 8; ++r) {
      const int row = bm + wave * 16 + r + half * 8;   // C layout: reg r -> M
      const int col = bn + nt * 16 + l16;
      float v = c[nt][r];
      if (bias) v += bias[col];
      if (relu) v = fmaxf(v, 0.f);
      const size_t idx = (size_t)row * N + col;
      if (resid) v += resid[idx];
      if (outF) outF[idx] = v;
      else      outB[idx] = f2bf(v);
    }
  }
}

// ---------------------------------------------------------------------------
// Flash-style causal attention; one wave = one 16-row Q tile of one (b,h).
// grid = (T/128, H, B), block = 256 (8 independent waves).
// ---------------------------------------------------------------------------
__global__ __launch_bounds__(256)
void attn_kernel(const uint16_t* __restrict__ qb, const uint16_t* __restrict__ kbp,
                 const uint16_t* __restrict__ vTb, uint16_t* __restrict__ attb) {
  __shared__ uint16_t psh[8][16][36];
  const int lane = threadIdx.x & 31, wave = threadIdx.x >> 5;
  const int half = lane >> 4, l16 = lane & 15;
  const int h = blockIdx.y, b = blockIdx.z;
  const int t0 = blockIdx.x * 128 + wave * 16;
  const size_t rowbase = (size_t)b * 1024;
  const int hbase = h * 64;
  const uint32_t* q32 = (const uint32_t*)qb;
  const uint32_t* k32 = (const uint32_t*)kbp;
  const uint32_t* v32 = (const uint32_t*)vTb;

  frag_ab qa0, qa1;  // head dims [0..31], [32..63]
  {
    const size_t qrow = (rowbase + t0 + l16) * 512 + (hbase >> 1);
    const int kb8 = half * 4;
    for (int i = 0; i < 4; ++i) {
      qa0.u[i]     = q32[qrow + kb8 + i];
      qa0.u[4 + i] = q32[qrow + 8 + kb8 + i];
      qa1.u[i]     = q32[qrow + 16 + kb8 + i];
      qa1.u[4 + i] = q32[qrow + 24 + kb8 + i];
    }
  }

  float rmax[8], rsum[8];
  v8f acc[4];
  for (int r = 0; r < 8; ++r) { rmax[r] = -1e30f; rsum[r] = 0.f; }
  for (int nt = 0; nt < 4; ++nt) acc[nt] = v8f_zero();

  const int nchunk = (t0 + 47) >> 5;
  for (int cix = 0; cix < nchunk; ++cix) {
    const int kb0 = cix * 32;
    if (cix + 1 < nchunk) {   // hint next K/V chunk toward the caches
      const int nb = kb0 + 32;
      __builtin_prefetch(&k32[(rowbase + nb + l16) * 512 + (hbase >> 1)], 0, 0);
      __builtin_prefetch(&v32[((size_t)(b * 16 + h) * 64 + l16) * 512 + (nb >> 1)], 0, 0);
    }
    frag_ab kf[4];
    for (int g = 0; g < 2; ++g) {
      const size_t krow = (rowbase + kb0 + g * 16 + l16) * 512 + (hbase >> 1);
      for (int i = 0; i < 8; ++i) {
        kf[g * 2 + 0].u[i] = k32[krow + (i >> 2) * 8 + half * 4 + (i & 3)];
        kf[g * 2 + 1].u[i] = k32[krow + 16 + (i >> 2) * 8 + half * 4 + (i & 3)];
      }
    }
    v8f c0 = v8f_zero(), c1 = v8f_zero();
    c0 = wmma_bf16(qa0, kf[0], c0); c0 = wmma_bf16(qa1, kf[1], c0);
    c1 = wmma_bf16(qa0, kf[2], c1); c1 = wmma_bf16(qa1, kf[3], c1);

    float p0[8], p1[8];
    for (int r = 0; r < 8; ++r) {
      const int row = t0 + r + half * 8;
      float s0 = c0[r] * 0.03125f;              // ref scales by D^-0.5 = 1/32
      float s1 = c1[r] * 0.03125f;
      if (kb0 + l16 > row)      s0 = -1e30f;    // causal mask
      if (kb0 + 16 + l16 > row) s1 = -1e30f;
      float tm = fmaxf(s0, s1);
      for (int m = 8; m >= 1; m >>= 1) tm = fmaxf(tm, __shfl_xor(tm, m, 16));
      const float nm = fmaxf(rmax[r], tm);
      const float alpha = __expf(rmax[r] - nm);
      rmax[r] = nm;
      s0 = __expf(s0 - nm); s1 = __expf(s1 - nm);
      float ps = s0 + s1;
      for (int m = 8; m >= 1; m >>= 1) ps += __shfl_xor(ps, m, 16);
      rsum[r] = rsum[r] * alpha + ps;
      for (int nt = 0; nt < 4; ++nt) acc[nt][r] *= alpha;
      p0[r] = s0; p1[r] = s1;
    }
    for (int r = 0; r < 8; ++r) {               // stage P (C-layout) as bf16
      const int m = r + half * 8;
      psh[wave][m][l16]      = f2bf(p0[r]);
      psh[wave][m][16 + l16] = f2bf(p1[r]);
    }
    frag_ab pa;                                 // reread in A-layout
    {
      const uint16_t* prow = &psh[wave][l16][0];
      const int kb8 = half * 8;
      for (int i = 0; i < 4; ++i) {
        pa.u[i]     = *(const uint32_t*)&prow[kb8 + 2 * i];
        pa.u[4 + i] = *(const uint32_t*)&prow[16 + kb8 + 2 * i];
      }
    }
    for (int nt = 0; nt < 4; ++nt) {            // P·V with transposed V
      frag_ab vf;
      const size_t vrow = ((size_t)(b * 16 + h) * 64 + nt * 16 + l16) * 512
                        + (kb0 >> 1) + half * 4;
      for (int i = 0; i < 8; ++i) vf.u[i] = v32[vrow + (i >> 2) * 8 + (i & 3)];
      acc[nt] = wmma_bf16(pa, vf, acc[nt]);
    }
  }

  for (int nt = 0; nt < 4; ++nt)
    for (int r = 0; r < 8; ++r) {
      const int row = t0 + r + half * 8;
      const float val = acc[nt][r] / rsum[r];
      attb[(rowbase + row) * 1024 + hbase + nt * 16 + l16] = f2bf(val);
    }
}

// ---------------------------------------------------------------------------
__global__ void zero_kernel(float* p) { if (threadIdx.x == 0) *p = 0.f; }

__global__ __launch_bounds__(256)
void loss_kernel(const float* __restrict__ logits, const int* __restrict__ tgt,
                 float* __restrict__ loss) {
  __shared__ float red[256];
  const int row = blockIdx.x, t = threadIdx.x;
  const float* lr = logits + (size_t)row * 32000;
  float mx = -1e30f;
  for (int i = t; i < 32000; i += 256) mx = fmaxf(mx, lr[i]);
  red[t] = mx; __syncthreads();
  for (int o = 128; o > 0; o >>= 1) { if (t < o) red[t] = fmaxf(red[t], red[t + o]); __syncthreads(); }
  mx = red[0]; __syncthreads();
  float s = 0.f;
  for (int i = t; i < 32000; i += 256) s += __expf(lr[i] - mx);
  red[t] = s; __syncthreads();
  for (int o = 128; o > 0; o >>= 1) { if (t < o) red[t] += red[t + o]; __syncthreads(); }
  if (t == 0) {
    const float lse = mx + __logf(red[0]);
    const float lp = lr[tgt[row]] - lse;
    atomicAdd(loss, -lp * (1.f / 4096.f));
  }
}

// ---------------------------------------------------------------------------
extern "C" void kernel_launch(void* const* d_in, const int* in_sizes, int n_in,
                              void* d_out, int out_size, void* d_ws, size_t ws_size,
                              hipStream_t stream) {
  const float* tok_emb = (const float*)d_in[0];
  const float* pos_emb = (const float*)d_in[1];
  const float* ln1_g   = (const float*)d_in[2];
  const float* ln1_b   = (const float*)d_in[3];
  const float* Wq      = (const float*)d_in[4];
  const float* Wk      = (const float*)d_in[5];
  const float* Wv      = (const float*)d_in[6];
  const float* Wo      = (const float*)d_in[7];
  const float* bo      = (const float*)d_in[8];
  const float* ln2_g   = (const float*)d_in[9];
  const float* ln2_b   = (const float*)d_in[10];
  const float* W1      = (const float*)d_in[11];
  const float* b1      = (const float*)d_in[12];
  const float* W2      = (const float*)d_in[13];
  const float* b2      = (const float*)d_in[14];
  const float* lnf_g   = (const float*)d_in[15];
  const float* lnf_b   = (const float*)d_in[16];
  const float* lm_W    = (const float*)d_in[17];
  const float* lm_b    = (const float*)d_in[18];
  const int*   ctx     = (const int*)d_in[19];
  const int*   tgt     = (const int*)d_in[20];

  float* logits = (float*)d_out;
  float* loss   = logits + (size_t)4096 * 32000;

  char* w = (char*)d_ws;
  float*    x    = (float*)w;    w += (size_t)4096 * 1024 * 4;   // 16 MB
  uint16_t* hb   = (uint16_t*)w; w += (size_t)4096 * 1024 * 2;   // 8 MB
  uint16_t* qb   = (uint16_t*)w; w += (size_t)4096 * 1024 * 2;
  uint16_t* kb   = (uint16_t*)w; w += (size_t)4096 * 1024 * 2;
  uint16_t* vb   = (uint16_t*)w; w += (size_t)4096 * 1024 * 2;
  uint16_t* vT   = (uint16_t*)w; w += (size_t)4096 * 1024 * 2;
  uint16_t* attb = (uint16_t*)w; w += (size_t)4096 * 1024 * 2;
  uint16_t* ff1  = (uint16_t*)w; w += (size_t)4096 * 4096 * 2;   // 32 MB
  uint16_t* wbuf = (uint16_t*)w;                                  // 64 MB, reused
  uint16_t* wq  = wbuf;
  uint16_t* wk  = wbuf + (size_t)1 * 1048576;
  uint16_t* wv  = wbuf + (size_t)2 * 1048576;
  uint16_t* wo  = wbuf + (size_t)3 * 1048576;
  uint16_t* w1  = wbuf + (size_t)4 * 1048576;
  uint16_t* w2  = wbuf + (size_t)8 * 1048576;
  uint16_t* wlm = wbuf;

  const dim3 blk(256);
  embed_kernel<<<4096, blk, 0, stream>>>(tok_emb, pos_emb, ctx, x);

  for (int l = 0; l < 8; ++l) {
    ln_kernel<<<4096, blk, 0, stream>>>(x, ln1_g + l * 1024, ln1_b + l * 1024, hb);

    cvt_qkv_t_kernel<<<4096, blk, 0, stream>>>(Wq + (size_t)l * 1048576, wq);
    cvt_qkv_t_kernel<<<4096, blk, 0, stream>>>(Wk + (size_t)l * 1048576, wk);
    cvt_qkv_t_kernel<<<4096, blk, 0, stream>>>(Wv + (size_t)l * 1048576, wv);
    cvt_t_kernel<<<4096, blk, 0, stream>>>(Wo + (size_t)l * 1048576, wo, 10, 1024);

    gemm_bf16_kernel<<<dim3(8, 32), blk, 0, stream>>>(hb, wq, nullptr, nullptr, nullptr, qb, 4096, 1024, 1024, 0);
    gemm_bf16_kernel<<<dim3(8, 32), blk, 0, stream>>>(hb, wk, nullptr, nullptr, nullptr, kb, 4096, 1024, 1024, 0);
    gemm_bf16_kernel<<<dim3(8, 32), blk, 0, stream>>>(hb, wv, nullptr, nullptr, nullptr, vb, 4096, 1024, 1024, 0);

    transpose_v_kernel<<<16384, blk, 0, stream>>>(vb, vT);
    attn_kernel<<<dim3(8, 16, 4), blk, 0, stream>>>(qb, kb, vT, attb);

    // x = x + att @ Wo + bo  (in-place residual)
    gemm_bf16_kernel<<<dim3(8, 32), blk, 0, stream>>>(attb, wo, bo + l * 1024, x, x, nullptr, 4096, 1024, 1024, 0);

    ln_kernel<<<4096, blk, 0, stream>>>(x, ln2_g + l * 1024, ln2_b + l * 1024, hb);
    cvt_t_kernel<<<16384, blk, 0, stream>>>(W1 + (size_t)l * 4194304, w1, 10, 4096);
    cvt_t_kernel<<<16384, blk, 0, stream>>>(W2 + (size_t)l * 4194304, w2, 12, 1024);

    gemm_bf16_kernel<<<dim3(32, 32), blk, 0, stream>>>(hb, w1, b1 + l * 4096, nullptr, nullptr, ff1, 4096, 4096, 1024, 1);
    gemm_bf16_kernel<<<dim3(8, 32), blk, 0, stream>>>(ff1, w2, b2 + l * 1024, x, x, nullptr, 4096, 1024, 4096, 0);
  }

  ln_kernel<<<4096, blk, 0, stream>>>(x, lnf_g, lnf_b, hb);
  cvt_t_kernel<<<128000, blk, 0, stream>>>(lm_W, wlm, 10, 32000);
  gemm_bf16_kernel<<<dim3(250, 32), blk, 0, stream>>>(hb, wlm, lm_b, nullptr, logits, nullptr, 4096, 32000, 1024, 0);

  zero_kernel<<<1, 32, 0, stream>>>(loss);
  loss_kernel<<<4096, blk, 0, stream>>>(logits, tgt, loss);
}
